// BENDR_77515569758294
// MI455X (gfx1250) — matrix operand order repaired and checked by reading, not compiled, not run on hardware
//
#include <hip/hip_runtime.h>
#include <hip/hip_bf16.h>
#include <stdint.h>

typedef __bf16 bf16_t;
typedef __attribute__((ext_vector_type(8)))  __bf16 v8bf;
typedef __attribute__((ext_vector_type(16))) __bf16 v16bf;
typedef __attribute__((ext_vector_type(8)))  float  v8f;

union Frag { v16bf v; v8bf h[2]; };

#define TILE_M 128
#define TILE_N 256
#define TILE_K 32
#define LDS_STRIDE 40   // halves: 80B rows, every 8-elem run stays 16B aligned

#if defined(__has_builtin)
#  if __has_builtin(__builtin_amdgcn_s_wait_asynccnt)
#    define WAIT_ASYNC(n) __builtin_amdgcn_s_wait_asynccnt(n)
#  endif
#endif
#ifndef WAIT_ASYNC
#  define WAIT_ASYNC(n) asm volatile("s_wait_asynccnt %0" :: "i"(n) : "memory")
#endif

// gfx1250 async global->LDS copy, 16B per lane, tracked by ASYNCcnt.
__device__ __forceinline__ void async_b128(unsigned lds_off, const void* gptr) {
  asm volatile("global_load_async_to_lds_b128 %0, %1, off"
               :: "v"(lds_off), "v"(gptr) : "memory");
}

// ---------------------------------------------------------------- small utils
__global__ __launch_bounds__(256) void k_f32_to_bf16(const float* __restrict__ in,
                                                     bf16_t* __restrict__ out, long n) {
  long i = (long)blockIdx.x * 256 + threadIdx.x;
  if (i < n) out[i] = (bf16_t)in[i];
}

__global__ __launch_bounds__(256) void k_dup(const float* __restrict__ in,
    float* __restrict__ of, bf16_t* __restrict__ ob, long n) {
  long i = (long)blockIdx.x * 256 + threadIdx.x;
  if (i < n) { float v = in[i]; of[i] = v; ob[i] = (bf16_t)v; }
}

__device__ inline float block_sum(float v, float* sh) {
  #pragma unroll
  for (int o = 16; o > 0; o >>= 1) v += __shfl_xor(v, o, 32);
  int lane = threadIdx.x & 31, wid = threadIdx.x >> 5;
  __syncthreads();
  if (lane == 0) sh[wid] = v;
  __syncthreads();
  v = (threadIdx.x < 8) ? sh[threadIdx.x] : 0.f;
  if (wid == 0) {
    #pragma unroll
    for (int o = 4; o > 0; o >>= 1) v += __shfl_xor(v, o, 32);
    if (lane == 0) sh[0] = v;
  }
  __syncthreads();
  return sh[0];
}

// ---------------------------------------------------------------- conv via im2col
// A[m][k]: m=b*Lout+lo, k=ci*3+t -> x[b][ci][2lo-1+t]; src either (B,Cin,L) or (B*L,Cin)
__global__ __launch_bounds__(256) void k_im2col(const float* __restrict__ src,
    bf16_t* __restrict__ A, int B, int Cin, int Lin, int Lout, int K3p, int src_is_BCL) {
  long idx = (long)blockIdx.x * 256 + threadIdx.x;
  long total = (long)B * Lout * K3p;
  if (idx >= total) return;
  int k = (int)(idx % K3p);
  long m = idx / K3p;
  int lo = (int)(m % Lout);
  int b  = (int)(m / Lout);
  bf16_t v = (bf16_t)0.0f;
  if (k < Cin * 3) {
    int ci = k / 3, t = k % 3;
    int li = 2 * lo - 1 + t;
    if (li >= 0 && li < Lin) {
      float f = src_is_BCL ? src[((long)b * Cin + ci) * Lin + li]
                           : src[((long)b * Lin + li) * Cin + ci];
      v = (bf16_t)f;
    }
  }
  A[idx] = v;
}

// BN stats over rows for column c of (Mrows, C) matrix
__global__ __launch_bounds__(256) void k_bn_stats_col(const float* __restrict__ y,
    float* __restrict__ mv, long Mrows, int C) {
  __shared__ float sh[8];
  int c = blockIdx.x;
  float s = 0.f, s2 = 0.f;
  for (long i = threadIdx.x; i < Mrows; i += 256) {
    float v = y[i * C + c];
    s += v; s2 += v * v;
  }
  float S  = block_sum(s,  sh);
  float S2 = block_sum(s2, sh);
  if (threadIdx.x == 0) {
    float m = S / (float)Mrows;
    mv[2 * c]     = m;
    mv[2 * c + 1] = S2 / (float)Mrows - m * m;
  }
}

__global__ __launch_bounds__(256) void k_bn_gelu_col(float* __restrict__ y,
    const float* __restrict__ mv, const float* __restrict__ g,
    const float* __restrict__ be, int C, long total) {
  long idx = (long)blockIdx.x * 256 + threadIdx.x;
  if (idx >= total) return;
  int c = (int)(idx % C);
  float m = mv[2 * c], v = mv[2 * c + 1];
  float t = g[c] * (y[idx] - m) * rsqrtf(v + 1e-5f) + be[c];
  y[idx] = 0.5f * t * (1.f + erff(t * 0.70710678118654752f));  // exact GELU
}

// ---------------------------------------------------------------- WMMA GEMM
// C[z] = A[z](MxK) * op(B[z]) + bias ; b_trans: B stored (N,K) else (K,N)
// double-buffered LDS; in-bounds tiles stream via global_load_async_to_lds_b128
__global__ __launch_bounds__(256) void k_gemm_bf16(
    const bf16_t* __restrict__ A, const bf16_t* __restrict__ B,
    const float* __restrict__ bias, float* __restrict__ Cf, bf16_t* __restrict__ Cb,
    int M, int N, int K, int lda, int ldb, int ldc,
    long sAb, long sAh, long sBb, long sBh, long sCb, long sCh,
    int nh, int b_trans, int relu) {
  __shared__ bf16_t As[2][TILE_M][LDS_STRIDE];
  __shared__ bf16_t Bs[2][TILE_N][LDS_STRIDE];

  int z  = blockIdx.z;
  int zb = z / nh, zh = z % nh;
  const bf16_t* Ab = A + zb * sAb + zh * sAh;
  const bf16_t* Bb = B + zb * sBb + zh * sBh;
  long cbase = zb * sCb + zh * sCh;

  int m0 = blockIdx.y * TILE_M;
  int n0 = blockIdx.x * TILE_N;
  int t = threadIdx.x;
  int lane = t & 31, wid = t >> 5;
  int wm = (wid & 1) * 64;        // 2 waves over M, 4 subtiles each
  int wn = (wid >> 1) * 64;       // 4 waves over N, 4 subtiles each
  int mfrag = lane & 15, khalf = lane >> 4;

  v8f acc[4][4] = {};

  auto stageA = [&](int buf, int k0) -> int {
    const bf16_t* base = Ab + (long)m0 * lda + k0;
    bool fast = (m0 + TILE_M <= M) && (k0 + TILE_K <= K) && ((lda & 7) == 0)
                && ((((size_t)base) & 15) == 0);
    if (fast) {
      #pragma unroll
      for (int c = t; c < TILE_M * 4; c += 256) {       // 2 chunks / thread
        int r = c >> 2, col = (c & 3) * 8;
        async_b128((unsigned)(uintptr_t)&As[buf][r][col], base + (long)r * lda + col);
      }
      return 2;
    }
    for (int c = t; c < TILE_M * 4; c += 256) {
      int r = c >> 2, col = (c & 3) * 8;
      int gm = m0 + r, gk = k0 + col;
      bf16_t* dst = &As[buf][r][col];
      #pragma unroll
      for (int j = 0; j < 8; ++j)
        dst[j] = (gm < M && gk + j < K) ? Ab[(long)gm * lda + gk + j] : (bf16_t)0.0f;
    }
    return 0;
  };

  auto stageB = [&](int buf, int k0) -> int {
    if (b_trans) {
      const bf16_t* base = Bb + (long)n0 * ldb + k0;
      bool fast = (n0 + TILE_N <= N) && (k0 + TILE_K <= K) && ((ldb & 7) == 0)
                  && ((((size_t)base) & 15) == 0);
      if (fast) {
        #pragma unroll
        for (int c = t; c < TILE_N * 4; c += 256) {     // 4 chunks / thread
          int r = c >> 2, col = (c & 3) * 8;
          async_b128((unsigned)(uintptr_t)&Bs[buf][r][col], base + (long)r * ldb + col);
        }
        return 4;
      }
      for (int c = t; c < TILE_N * 4; c += 256) {
        int r = c >> 2, col = (c & 3) * 8;
        int gn = n0 + r, gk = k0 + col;
        bf16_t* dst = &Bs[buf][r][col];
        #pragma unroll
        for (int j = 0; j < 8; ++j)
          dst[j] = (gn < N && gk + j < K) ? Bb[(long)gn * ldb + gk + j] : (bf16_t)0.0f;
      }
      return 0;
    }
    // (K,N) storage: transposing gather, synchronous
    for (int c = t; c < TILE_N * TILE_K; c += 256) {
      int nn = c & 255, kk = c >> 8;
      int gn = n0 + nn, gk = k0 + kk;
      Bs[buf][nn][kk] = (gn < N && gk < K) ? Bb[(long)gk * ldb + gn] : (bf16_t)0.0f;
    }
    return 0;
  };

  int nsteps = (K + TILE_K - 1) / TILE_K;
  int cur = 0;
  (void)stageA(0, 0);
  (void)stageB(0, 0);

  for (int s = 0; s < nsteps; ++s) {
    int nn = 0;
    if (s + 1 < nsteps) {
      nn += stageA(cur ^ 1, (s + 1) * TILE_K);
      nn += stageB(cur ^ 1, (s + 1) * TILE_K);
    }
    // async completes in-order: waiting "<= just-issued" drains the current buffer
    if      (nn == 0) WAIT_ASYNC(0);
    else if (nn == 2) WAIT_ASYNC(2);
    else if (nn == 4) WAIT_ASYNC(4);
    else              WAIT_ASYNC(6);
    __syncthreads();

    Frag af[4], bfr[4];
    #pragma unroll
    for (int i = 0; i < 4; ++i) {
      int row = wm + i * 16 + mfrag;
      af[i].h[0] = *(const v8bf*)&As[cur][row][khalf * 8];       // K = kh*8..+7
      af[i].h[1] = *(const v8bf*)&As[cur][row][16 + khalf * 8];  // K = 16+kh*8..+7
    }
    #pragma unroll
    for (int j = 0; j < 4; ++j) {
      int rn = wn + j * 16 + mfrag;
      bfr[j].h[0] = *(const v8bf*)&Bs[cur][rn][khalf * 8];       // K = kh*8..+7
      bfr[j].h[1] = *(const v8bf*)&Bs[cur][rn][16 + khalf * 8];  // K = 16+kh*8..+7
    }
    #pragma unroll
    for (int i = 0; i < 4; ++i)
      #pragma unroll
      for (int j = 0; j < 4; ++j)
        acc[i][j] = __builtin_amdgcn_wmma_f32_16x16x32_bf16(
            false, af[i].v, false, bfr[j].v, (short)0, acc[i][j], false, false);
    __syncthreads();
    cur ^= 1;
  }

  // epilogue: C VGPR r -> (M = r + 8*(lane>=16), N = lane&15)
  int nlo = lane & 15, mhi = lane >> 4;
  #pragma unroll
  for (int i = 0; i < 4; ++i)
    #pragma unroll
    for (int j = 0; j < 4; ++j)
      #pragma unroll
      for (int r = 0; r < 8; ++r) {
        int gm = m0 + wm + i * 16 + r + 8 * mhi;
        int gn = n0 + wn + j * 16 + nlo;
        if (gm < M && gn < N) {
          float vv = acc[i][j][r];
          if (bias) vv += bias[gn];
          if (relu) vv = fmaxf(vv, 0.f);
          long idx = cbase + (long)gm * ldc + gn;
          if (Cf) Cf[idx] = vv;
          if (Cb) Cb[idx] = (bf16_t)vv;
        }
      }
}

// ---------------------------------------------------------------- softmax (wave per row)
__global__ __launch_bounds__(256) void k_softmax(const float* __restrict__ E,
    bf16_t* __restrict__ P, int S, float scale, long nrows) {
  int wid = threadIdx.x >> 5, lane = threadIdx.x & 31;
  long row = (long)blockIdx.x * 8 + wid;
  if (row >= nrows) return;
  const float* e = E + row * S;
  float mx = -3.4e38f;
  for (int j = lane; j < S; j += 32) mx = fmaxf(mx, e[j] * scale);
  #pragma unroll
  for (int o = 16; o > 0; o >>= 1) mx = fmaxf(mx, __shfl_xor(mx, o, 32));
  float ex[8]; int cnt = 0; float sum = 0.f;
  for (int j = lane; j < S; j += 32) { float u = expf(e[j] * scale - mx); ex[cnt++] = u; sum += u; }
  #pragma unroll
  for (int o = 16; o > 0; o >>= 1) sum += __shfl_xor(sum, o, 32);
  float inv = 1.f / sum;
  cnt = 0;
  for (int j = lane; j < S; j += 32) P[row * S + j] = (bf16_t)(ex[cnt++] * inv);
}

// ---------------------------------------------------------------- residual + LayerNorm
__global__ __launch_bounds__(256) void k_add_ln(const float* __restrict__ a,
    const float* __restrict__ res, const float* __restrict__ g, const float* __restrict__ be,
    float* __restrict__ of, bf16_t* __restrict__ ob, int E) {
  __shared__ float sh[8];
  long row = blockIdx.x;
  const float* pa = a + row * E;
  const float* pr = res + row * E;
  float loc[8]; int cnt = 0; float s = 0.f;
  for (int j = threadIdx.x; j < E; j += 256) { float v = pa[j] + pr[j]; loc[cnt++] = v; s += v; }
  float mean = block_sum(s, sh) / (float)E;
  float s2 = 0.f;
  for (int i = 0; i < cnt; ++i) { float d = loc[i] - mean; s2 += d * d; }
  float var = block_sum(s2, sh) / (float)E;
  float rinv = rsqrtf(var + 1e-5f);
  cnt = 0;
  for (int j = threadIdx.x; j < E; j += 256) {
    float v = (loc[cnt++] - mean) * rinv * g[j] + be[j];
    of[row * E + j] = v;
    ob[row * E + j] = (bf16_t)v;
  }
}

// ---------------------------------------------------------------- host orchestration
static inline void gemm(hipStream_t s, const bf16_t* A, const bf16_t* B, const float* bias,
                        float* Cf, bf16_t* Cb, int M, int N, int K,
                        int lda, int ldb, int ldc,
                        long sAb, long sAh, long sBb, long sBh, long sCb, long sCh,
                        int nh, int Z, int b_trans, int relu) {
  dim3 g((N + TILE_N - 1) / TILE_N, (M + TILE_M - 1) / TILE_M, Z);
  k_gemm_bf16<<<g, 256, 0, s>>>(A, B, bias, Cf, Cb, M, N, K, lda, ldb, ldc,
                                sAb, sAh, sBb, sBh, sCb, sCh, nh, b_trans, relu);
}

extern "C" void kernel_launch(void* const* d_in, const int* in_sizes, int n_in,
                              void* d_out, int out_size, void* d_ws, size_t ws_size,
                              hipStream_t stream) {
  const int Bn = 32, S = 256, E = 1536, F = 3076, H = 8, HD = 192;
  const long BSE  = (long)Bn * S * E;       // 12,582,912
  const long BHSS = (long)Bn * H * S * S;   // 16,777,216

  char* p = (char*)d_ws;
  auto alloc = [&](size_t bytes) { char* r = p; p += (bytes + 255) & ~(size_t)255; return r; };

  float*  convA  = (float*)alloc(67108864ULL * 4);       // ping (stage 0/2/4 outputs)
  float*  convB  = (float*)alloc(33554432ULL * 4);       // pong (stage 1/3 outputs)
  bf16_t* imBuf  = (bf16_t*)alloc(65536ULL * 1536 * 2);  // im2col scratch (max stage 1)
  float*  mv     = (float*)alloc(2 * 1536 * 4);
  float*  Xf     = (float*)alloc(BSE * 4);
  bf16_t* Xb     = (bf16_t*)alloc(BSE * 2);
  bf16_t* Qb     = (bf16_t*)alloc(BSE * 2);
  bf16_t* Kb     = (bf16_t*)alloc(BSE * 2);
  bf16_t* Vb     = (bf16_t*)alloc(BSE * 2);
  bf16_t* Ob     = (bf16_t*)alloc(BSE * 2);
  float*  energy = (float*)alloc(BHSS * 4);
  bf16_t* Pb     = (bf16_t*)alloc(BHSS * 2);
  float*  OPf    = (float*)alloc(BSE * 4);
  float*  Hf     = (float*)alloc(BSE * 4);
  bf16_t* Hb     = (bf16_t*)alloc(BSE * 2);
  const int F1LD = 3080;                                 // pad 3076 rows to 16B multiple
  bf16_t* F1b    = (bf16_t*)alloc(8192ULL * F1LD * 2);
  float*  Ff     = (float*)alloc(BSE * 4);
  bf16_t* wqb    = (bf16_t*)alloc((size_t)E * E * 2);
  bf16_t* wkb    = (bf16_t*)alloc((size_t)E * E * 2);
  bf16_t* wvb    = (bf16_t*)alloc((size_t)E * E * 2);
  bf16_t* wob    = (bf16_t*)alloc((size_t)E * E * 2);
  bf16_t* w1b    = (bf16_t*)alloc((size_t)F * E * 2);
  bf16_t* w2b    = (bf16_t*)alloc((size_t)F * E * 2);

  // ---- conv tower as im2col + WMMA GEMM (+ column BN/GELU)
  struct St { int ci, co, li, lo, k3p; };
  St st[5] = {{14,512,8192,4096,48},{512,512,4096,2048,1536},{512,512,2048,1024,1536},
              {512,512,1024,512,1536},{512,1536,512,256,1536}};
  const float* src = (const float*)d_in[0];
  float* dsts[5] = {convA, convB, convA, convB, convA};
  for (int i = 0; i < 5; ++i) {
    const float* cw = (const float*)d_in[1 + 4 * i];
    const float* cb = (const float*)d_in[2 + 4 * i];
    const float* gg = (const float*)d_in[3 + 4 * i];
    const float* be = (const float*)d_in[4 + 4 * i];
    float* dst = dsts[i];
    long Mr = (long)Bn * st[i].lo;
    int K3 = st[i].ci * 3;
    long imTot = Mr * st[i].k3p;
    k_im2col<<<(imTot + 255) / 256, 256, 0, stream>>>(src, imBuf, Bn, st[i].ci, st[i].li,
                                                      st[i].lo, st[i].k3p, i == 0 ? 1 : 0);
    long wTot = (long)st[i].co * K3;
    k_f32_to_bf16<<<(wTot + 255) / 256, 256, 0, stream>>>(cw, wqb, wTot);
    gemm(stream, imBuf, wqb, cb, dst, nullptr, (int)Mr, st[i].co, K3,
         st[i].k3p, K3, st[i].co, 0,0,0,0,0,0, 1, 1, 1, 0);
    k_bn_stats_col<<<st[i].co, 256, 0, stream>>>(dst, mv, Mr, st[i].co);
    long tot = Mr * st[i].co;
    k_bn_gelu_col<<<(tot + 255) / 256, 256, 0, stream>>>(dst, mv, gg, be, st[i].co, tot);
    src = dst;
  }
  // stage-4 output is already (B, S, E) row-major
  k_dup<<<(BSE + 255) / 256, 256, 0, stream>>>(dsts[4], Xf, Xb, BSE);

  // ---- transformer blocks
  const float* wq  = (const float*)d_in[21];
  const float* wk  = (const float*)d_in[22];
  const float* wv  = (const float*)d_in[23];
  const float* wo  = (const float*)d_in[24];
  const float* w1  = (const float*)d_in[25];
  const float* b1  = (const float*)d_in[26];
  const float* w2  = (const float*)d_in[27];
  const float* b2  = (const float*)d_in[28];
  const float* l1g = (const float*)d_in[29];
  const float* l1b = (const float*)d_in[30];
  const float* l2g = (const float*)d_in[31];
  const float* l2b = (const float*)d_in[32];

  const long EEn = (long)E * E, FEn = (long)F * E;
  const int M = Bn * S;                       // 8192
  const float scl = 0.07216878364870323f;     // 1/sqrt(192)

  for (int l = 0; l < 8; ++l) {
    k_f32_to_bf16<<<(EEn + 255) / 256, 256, 0, stream>>>(wq + l * EEn, wqb, EEn);
    k_f32_to_bf16<<<(EEn + 255) / 256, 256, 0, stream>>>(wk + l * EEn, wkb, EEn);
    k_f32_to_bf16<<<(EEn + 255) / 256, 256, 0, stream>>>(wv + l * EEn, wvb, EEn);
    k_f32_to_bf16<<<(EEn + 255) / 256, 256, 0, stream>>>(wo + l * EEn, wob, EEn);
    k_f32_to_bf16<<<(FEn + 255) / 256, 256, 0, stream>>>(w1 + l * FEn, w1b, FEn);
    k_f32_to_bf16<<<(FEn + 255) / 256, 256, 0, stream>>>(w2 + l * FEn, w2b, FEn);

    // Q,K,V = X @ W^T  ((b,s,h,d) layout, bf16 outputs)
    gemm(stream, Xb, wqb, nullptr, nullptr, Qb, M, E, E, E, E, E, 0,0,0,0,0,0, 1, 1, 1, 0);
    gemm(stream, Xb, wkb, nullptr, nullptr, Kb, M, E, E, E, E, E, 0,0,0,0,0,0, 1, 1, 1, 0);
    gemm(stream, Xb, wvb, nullptr, nullptr, Vb, M, E, E, E, E, E, 0,0,0,0,0,0, 1, 1, 1, 0);

    // energy[z] = Q_bh @ K_bh^T  (z = b*8 + h)
    gemm(stream, Qb, Kb, nullptr, energy, nullptr, S, S, HD, E, E, S,
         (long)S * E, HD, (long)S * E, HD, 8L * S * S, (long)S * S, H, Bn * H, 1, 0);
    k_softmax<<<(Bn * H * S) / 8, 256, 0, stream>>>(energy, Pb, S, scl, (long)Bn * H * S);

    // O_bh = P @ V  (B stored (K=S, N=HD) row-major -> b_trans=0)
    gemm(stream, Pb, Vb, nullptr, nullptr, Ob, S, HD, S, S, E, E,
         8L * S * S, (long)S * S, (long)S * E, HD, (long)S * E, HD, H, Bn * H, 0, 0);

    // out-proj, residual + LN1
    gemm(stream, Ob, wob, nullptr, OPf, nullptr, M, E, E, E, E, E, 0,0,0,0,0,0, 1, 1, 1, 0);
    k_add_ln<<<M, 256, 0, stream>>>(OPf, Xf, l1g + l * E, l1b + l * E, Hf, Hb, E);

    // FFN: relu(H @ W1^T + b1) @ W2^T + b2, residual + LN2
    gemm(stream, Hb, w1b, b1 + (long)l * F, nullptr, F1b, M, F, E, E, E, F1LD,
         0,0,0,0,0,0, 1, 1, 1, 1);
    gemm(stream, F1b, w2b, b2 + (long)l * E, Ff, nullptr, M, E, F, F1LD, F, E,
         0,0,0,0,0,0, 1, 1, 1, 0);
    k_add_ln<<<M, 256, 0, stream>>>(Ff, Hf, l2g + l * E, l2b + l * E, Xf, Xb, E);
  }

  hipMemcpyAsync(d_out, Xf, (size_t)out_size * sizeof(float),
                 hipMemcpyDeviceToDevice, stream);
  (void)in_sizes; (void)n_in; (void)ws_size;
}
